// MultiheadAttention_4904852652122
// MI455X (gfx1250) — matrix operand order repaired and checked by reading, not compiled, not run on hardware
//
#include <hip/hip_runtime.h>

// ---------------------------------------------------------------------------
// MI455X / gfx1250 multi-head attention forward.
// All GEMMs use v_wmma_f32_16x16x32_bf16 (wave32, 16x16 tiles, K=32).
// ---------------------------------------------------------------------------

typedef __attribute__((ext_vector_type(16))) __bf16 v16bf;
typedef __attribute__((ext_vector_type(8)))  __bf16 v8bf;
typedef __attribute__((ext_vector_type(8)))  float  v8f;

#define TLEN   1024
#define BSZ    4
#define EMB    512
#define HEADS  16
#define HD     32
#define BH     (BSZ * HEADS)     // 64
#define ROWS   (TLEN * BSZ)      // 4096

// ---- WMMA fragment loads (bf16, 16x16x32) ---------------------------------
// A-matrix 16x32: lane<16 -> M=lane,  K = 0..7 and 16..23
//                 lane>=16 -> M=lane-16, K = 8..15 and 24..31
__device__ __forceinline__ v16bf load_frag_a(const __bf16* base, int ld) {
  int lane = threadIdx.x & 31;
  int m  = lane & 15;
  int kh = (lane & 16) ? 8 : 0;
  const __bf16* p = base + (size_t)m * ld + kh;
  v8bf lo = *(const v8bf*)(p);
  v8bf hi = *(const v8bf*)(p + 16);
  v16bf r;
#pragma unroll
  for (int i = 0; i < 8; ++i) { r[i] = lo[i]; r[i + 8] = hi[i]; }
  return r;
}

// B-matrix 32x16 supplied as B^T row-major (row n == column n of B):
// lane<16 -> N=lane,  K=0..7 and 16..23; lane>=16 -> N=lane-16, K=8..15 and 24..31
__device__ __forceinline__ v16bf load_frag_b(const __bf16* base, int ld) {
  int lane = threadIdx.x & 31;
  int n  = lane & 15;
  int kh = (lane & 16) ? 8 : 0;
  const __bf16* p = base + (size_t)n * ld + kh;
  v8bf lo = *(const v8bf*)(p);
  v8bf hi = *(const v8bf*)(p + 16);
  v16bf r;
#pragma unroll
  for (int i = 0; i < 8; ++i) { r[i] = lo[i]; r[i + 8] = hi[i]; }
  return r;
}

__device__ __forceinline__ v8f wmma_bf16(v16bf a, v16bf b, v8f c) {
  return __builtin_amdgcn_wmma_f32_16x16x32_bf16(
      false, a, false, b, (short)0, c, false, false);
}

// ---- f32 -> bf16 conversion -----------------------------------------------
__global__ __launch_bounds__(256) void cvt_bf16(const float* __restrict__ in,
                                                __bf16* __restrict__ out, int n) {
  int i = blockIdx.x * 256 + threadIdx.x;
  if (i < n) out[i] = (__bf16)in[i];
}

// ---- projection / output GEMM: Y = (X @ W^T + bias) * scale ---------------
// X: [4096, 512] bf16 row-major, W: [512, 512] bf16 row-major (used as B^T).
// mode 0: Y -> bf16 flat [4096,512]        (Q / K projections)
// mode 1: Y -> bf16 head-transposed V^T: [BH, HD, TLEN]
// mode 2: Y -> f32 flat [4096,512]         (final output projection)
__global__ __launch_bounds__(256) void gemm_proj(
    const __bf16* __restrict__ X, const __bf16* __restrict__ W,
    const float* __restrict__ bias, float scale, int mode,
    __bf16* __restrict__ outb, float* __restrict__ outf) {
  int lane = threadIdx.x & 31;
  int w    = threadIdx.x >> 5;               // 8 waves
  int row0 = blockIdx.x * 32 + (w >> 2) * 16;
  int col0 = blockIdx.y * 64 + (w & 3) * 16;
  v8f acc = {};
#pragma unroll 4
  for (int k = 0; k < EMB; k += 32) {
    const __bf16* ap = X + (size_t)row0 * EMB + k;
    const __bf16* bp = W + (size_t)col0 * EMB + k;
    if (k + 32 < EMB) {
      __builtin_prefetch(ap + 32, 0, 1);
      __builtin_prefetch(bp + 32, 0, 1);
    }
    v16bf a = load_frag_a(ap, EMB);
    v16bf b = load_frag_b(bp, EMB);
    acc = wmma_bf16(a, b, acc);
  }
  int nl   = lane & 15;
  int mofs = (lane & 16) ? 8 : 0;
#pragma unroll
  for (int j = 0; j < 8; ++j) {
    int m = row0 + j + mofs;
    int n = col0 + nl;
    float v = (acc[j] + bias[n]) * scale;
    if (mode == 0) {
      outb[(size_t)m * EMB + n] = (__bf16)v;
    } else if (mode == 1) {
      int t = m >> 2, b = m & 3, h = n >> 5, d = n & 31;
      outb[((size_t)(b * HEADS + h) * HD + d) * TLEN + t] = (__bf16)v;
    } else {
      outf[(size_t)m * EMB + n] = v;
    }
  }
}

// ---- fused scores + bias + mask + softmax ---------------------------------
// One block = 16 query rows of one head. Q/K in flat [4096,512] bf16 layout
// (row stride 2048 elems, head slice contiguous 32 = one K-step).
// Score strip (16 x 1024 f32 = 64KB) lives in LDS; softmax via shfl.
__global__ __launch_bounds__(256) void attn_scores_softmax(
    const __bf16* __restrict__ Q, const __bf16* __restrict__ K,
    const float* __restrict__ bias, const int* __restrict__ kpm,
    __bf16* __restrict__ probs) {
  __shared__ float S[16][TLEN];
  int bh   = blockIdx.x;           // 0..63, bh = b*16 + h
  int b    = bh >> 4;
  int h    = bh & 15;
  int row0 = blockIdx.y * 16;
  int lane = threadIdx.x & 31;
  int w    = threadIdx.x >> 5;

  const __bf16* Qbase = Q + ((size_t)row0 * BSZ + b) * EMB + h * HD;
  v16bf a = load_frag_a(Qbase, BSZ * EMB);   // Q rows row0..row0+15, K=0..31

  int nl   = lane & 15;
  int mofs = (lane & 16) ? 8 : 0;
#pragma unroll 2
  for (int ci = 0; ci < 8; ++ci) {
    int colbase = ci * 128 + w * 16;
    const __bf16* Kbase = K + ((size_t)colbase * BSZ + b) * EMB + h * HD;
    v16bf bb = load_frag_b(Kbase, BSZ * EMB);
    v8f acc = {};
    acc = wmma_bf16(a, bb, acc);
    int n = colbase + nl;
    int masked = kpm[b * TLEN + n];
#pragma unroll
    for (int j = 0; j < 8; ++j) {
      int m = j + mofs;
      float sv = acc[j] + bias[((size_t)bh * TLEN + row0 + m) * TLEN + n];
      S[m][n] = masked ? -__builtin_inff() : sv;
    }
  }
  __syncthreads();

  // softmax: each row handled by 16 lanes of one wave half
  int r    = threadIdx.x >> 4;     // 0..15
  int cgrp = threadIdx.x & 15;
  float mx = -__builtin_inff();
  for (int c = cgrp; c < TLEN; c += 16) mx = fmaxf(mx, S[r][c]);
#pragma unroll
  for (int off = 8; off >= 1; off >>= 1) mx = fmaxf(mx, __shfl_xor(mx, off, 32));
  float sum = 0.f;
  for (int c = cgrp; c < TLEN; c += 16) {
    float e = __expf(S[r][c] - mx);
    S[r][c] = e;
    sum += e;
  }
#pragma unroll
  for (int off = 8; off >= 1; off >>= 1) sum += __shfl_xor(sum, off, 32);
  float inv = 1.0f / sum;
  __bf16* prow = probs + ((size_t)bh * TLEN + row0 + r) * TLEN;
  for (int c = cgrp; c < TLEN; c += 16) prow[c] = (__bf16)(S[r][c] * inv);
}

// ---- attn = probs @ V  (per head, V pre-transposed [BH, HD, TLEN]) --------
__global__ __launch_bounds__(256) void attn_pv(
    const __bf16* __restrict__ probs, const __bf16* __restrict__ vT,
    __bf16* __restrict__ attnb) {
  int bh   = blockIdx.x;
  int lane = threadIdx.x & 31;
  int w    = threadIdx.x >> 5;
  int row0 = blockIdx.y * 64 + (w >> 1) * 16;
  int col0 = (w & 1) * 16;
  const __bf16* P = probs + (size_t)bh * TLEN * TLEN;
  const __bf16* V = vT + (size_t)bh * HD * TLEN;
  v8f acc = {};
#pragma unroll 4
  for (int k = 0; k < TLEN; k += 32) {
    const __bf16* ap = P + (size_t)row0 * TLEN + k;
    const __bf16* bp = V + (size_t)col0 * TLEN + k;
    if (k + 32 < TLEN) {
      __builtin_prefetch(ap + 32, 0, 1);
      __builtin_prefetch(bp + 32, 0, 1);
    }
    v16bf a  = load_frag_a(ap, TLEN);
    v16bf bb = load_frag_b(bp, TLEN);
    acc = wmma_bf16(a, bb, acc);
  }
  int b = bh >> 4, h = bh & 15;
  int nl   = lane & 15;
  int mofs = (lane & 16) ? 8 : 0;
#pragma unroll
  for (int j = 0; j < 8; ++j) {
    int t = row0 + j + mofs;
    int n = col0 + nl;
    attnb[((size_t)t * BSZ + b) * EMB + h * HD + n] = (__bf16)acc[j];
  }
}

// ---- head-averaged attention weights --------------------------------------
__global__ __launch_bounds__(256) void avg_probs(
    const __bf16* __restrict__ probs, float* __restrict__ outavg) {
  size_t idx = (size_t)blockIdx.x * 256 + threadIdx.x;
  if (idx >= (size_t)BSZ * TLEN * TLEN) return;
  size_t b  = idx >> 20;                 // / (1024*1024)
  size_t ts = idx & ((1u << 20) - 1);
  const __bf16* p = probs + (b * HEADS) * (size_t)(TLEN * TLEN) + ts;
  float s = 0.f;
#pragma unroll
  for (int h = 0; h < HEADS; ++h) s += (float)p[(size_t)h * TLEN * TLEN];
  outavg[idx] = s * (1.0f / HEADS);
}

// ---------------------------------------------------------------------------
extern "C" void kernel_launch(void* const* d_in, const int* in_sizes, int n_in,
                              void* d_out, int out_size, void* d_ws, size_t ws_size,
                              hipStream_t stream) {
  const float* query = (const float*)d_in[0];
  const float* attn_bias = (const float*)d_in[1];
  const int*   kpm   = (const int*)d_in[2];
  const float* q_w = (const float*)d_in[3];
  const float* q_b = (const float*)d_in[4];
  const float* k_w = (const float*)d_in[5];
  const float* k_b = (const float*)d_in[6];
  const float* v_w = (const float*)d_in[7];
  const float* v_b = (const float*)d_in[8];
  const float* out_w = (const float*)d_in[9];
  const float* out_b = (const float*)d_in[10];

  float* out_f   = (float*)d_out;                        // [T,B,E] = 2,097,152
  float* out_avg = out_f + (size_t)TLEN * BSZ * EMB;     // [B,T,T] = 4,194,304

  // workspace layout (bf16 buffers)
  char* ws = (char*)d_ws;
  __bf16* x_bf   = (__bf16*)(ws);                              // 4 MB
  __bf16* qw_bf  = (__bf16*)(ws + 4194304);                    // 512 KB
  __bf16* kw_bf  = (__bf16*)(ws + 4718592);
  __bf16* vw_bf  = (__bf16*)(ws + 5242880);
  __bf16* ow_bf  = (__bf16*)(ws + 5767168);
  __bf16* q_bf   = (__bf16*)(ws + 6291456);                    // 4 MB
  __bf16* k_bf   = (__bf16*)(ws + 10485760);                   // 4 MB
  __bf16* vT_bf  = (__bf16*)(ws + 14680064);                   // 4 MB
  __bf16* at_bf  = (__bf16*)(ws + 18874368);                   // 4 MB
  __bf16* probs  = (__bf16*)(ws + 23068672);                   // 128 MB

  const int NX = ROWS * EMB;      // 2,097,152
  const int NW = EMB * EMB;       // 262,144
  cvt_bf16<<<(NX + 255) / 256, 256, 0, stream>>>(query, x_bf, NX);
  cvt_bf16<<<(NW + 255) / 256, 256, 0, stream>>>(q_w, qw_bf, NW);
  cvt_bf16<<<(NW + 255) / 256, 256, 0, stream>>>(k_w, kw_bf, NW);
  cvt_bf16<<<(NW + 255) / 256, 256, 0, stream>>>(v_w, vw_bf, NW);
  cvt_bf16<<<(NW + 255) / 256, 256, 0, stream>>>(out_w, ow_bf, NW);

  dim3 ggrid(ROWS / 32, EMB / 64);
  const float scaling = 0.17677669529663687f;  // 32^-0.5
  gemm_proj<<<ggrid, 256, 0, stream>>>(x_bf, qw_bf, q_b, scaling, 0, q_bf, nullptr);
  gemm_proj<<<ggrid, 256, 0, stream>>>(x_bf, kw_bf, k_b, 1.0f, 0, k_bf, nullptr);
  gemm_proj<<<ggrid, 256, 0, stream>>>(x_bf, vw_bf, v_b, 1.0f, 1, vT_bf, nullptr);

  attn_scores_softmax<<<dim3(BH, TLEN / 16), 256, 0, stream>>>(
      q_bf, k_bf, attn_bias, kpm, probs);

  attn_pv<<<dim3(BH, TLEN / 64), 256, 0, stream>>>(probs, vT_bf, at_bf);

  gemm_proj<<<ggrid, 256, 0, stream>>>(at_bf, ow_bf, out_b, 1.0f, 2, nullptr, out_f);

  avg_probs<<<(BSZ * TLEN * TLEN + 255) / 256, 256, 0, stream>>>(probs, out_avg);
}
